// MoEBlock_86620900426230
// MI455X (gfx1250) — matrix operand order repaired and checked by reading, not compiled, not run on hardware
//
#include <hip/hip_runtime.h>
#include <hip/hip_bf16.h>

// ---------------- types for WMMA fragments (wave32, gfx1250) ----------------
typedef __attribute__((ext_vector_type(16))) __bf16 v16bf;
typedef __attribute__((ext_vector_type(8)))  float  v8f;

#define BATCH 16
#define CCH   192           // channels (in == out)
#define HWSZ  1024          // 32*32
#define PAD   34            // zero-padded spatial extent
#define NEXP  8
#define KTOT  1728          // 9 * 192
#define NSTEP 54            // KTOT / 32

// ---------------------------------------------------------------------------
// Kernel 1: router. One block per sample. pooled -> gate -> softmax -> top2.
// ---------------------------------------------------------------------------
__global__ __launch_bounds__(192) void moe_router(
    const float* __restrict__ x, const float* __restrict__ gate_w,
    const float* __restrict__ gate_b, float* __restrict__ combine)
{
    __shared__ float pooled[CCH];
    __shared__ float logit[NEXP];
    const int b = blockIdx.x;
    const int t = threadIdx.x;

    if (t < CCH) {
        const float* p = x + ((size_t)b * CCH + t) * HWSZ;
        float s = 0.f;
        for (int k = 0; k < HWSZ; ++k) s += p[k];
        pooled[t] = s * (1.0f / (float)HWSZ);
    }
    __syncthreads();
    if (t < NEXP) {
        float s = gate_b[t];
        for (int c = 0; c < CCH; ++c) s += pooled[c] * gate_w[c * NEXP + t];
        logit[t] = s;
    }
    __syncthreads();
    if (t == 0) {
        float mx = logit[0];
        for (int e = 1; e < NEXP; ++e) mx = fmaxf(mx, logit[e]);
        float w[NEXP]; float ss = 0.f;
        for (int e = 0; e < NEXP; ++e) { w[e] = expf(logit[e] - mx); ss += w[e]; }
        const float inv = 1.0f / ss;
        for (int e = 0; e < NEXP; ++e) w[e] *= inv;
        int i0 = 0;                                  // top-2, earliest on ties
        for (int e = 1; e < NEXP; ++e) if (w[e] > w[i0]) i0 = e;
        int i1 = (i0 == 0) ? 1 : 0;
        for (int e = 0; e < NEXP; ++e) if (e != i0 && w[e] > w[i1]) i1 = e;
        const float s2 = w[i0] + w[i1] + 1e-10f;
        for (int e = 0; e < NEXP; ++e) combine[b * NEXP + e] = 0.f;
        combine[b * NEXP + i0] = w[i0] / s2;
        combine[b * NEXP + i1] = w[i1] / s2;
    }
}

// ---------------------------------------------------------------------------
// Kernel 2: x (NCHW f32) -> zero-padded NHWC bf16 [B][34][34][192].
// ---------------------------------------------------------------------------
__global__ __launch_bounds__(256) void moe_xpad(
    const float* __restrict__ x, __bf16* __restrict__ xp)
{
    const int idx = blockIdx.x * 256 + threadIdx.x;
    if (idx >= BATCH * PAD * PAD * CCH) return;
    const int c  = idx % CCH;
    const int t  = idx / CCH;
    const int pw = t % PAD;
    const int ph = (t / PAD) % PAD;
    const int b  = t / (PAD * PAD);
    const int h = ph - 1, w = pw - 1;
    float v = 0.f;
    if ((unsigned)h < 32u && (unsigned)w < 32u)
        v = x[((size_t)b * CCH + c) * HWSZ + h * 32 + w];
    xp[idx] = (__bf16)v;
}

// ---------------------------------------------------------------------------
// Kernel 3: combined per-sample weights -> wc[b][o][r][i'] bf16, i' carrying
// the per-32-block map pi (identity: contiguous k) so a B fragment
// is 16 contiguous bf16 matching A's contiguous-K layout (k = 16*hl + s).
// ---------------------------------------------------------------------------
__global__ __launch_bounds__(256) void moe_wcomb(
    const float* __restrict__ expert_w, const float* __restrict__ combine,
    __bf16* __restrict__ wc)
{
    const int t = blockIdx.x * 256 + threadIdx.x;
    if (t >= BATCH * CCH * CCH) return;
    const int i = t % CCH;
    const int o = (t / CCH) % CCH;
    const int b = t / (CCH * CCH);

    // pi: identity (contiguous k within each 32-block)
    const int j  = i & 31;
    const int jp = j;
    const int islot = (i & ~31) | jp;

    float acc[9];
#pragma unroll
    for (int r = 0; r < 9; ++r) acc[r] = 0.f;

    for (int e = 0; e < NEXP; ++e) {
        const float cw = combine[b * NEXP + e];
        if (cw != 0.0f) {
            const float* p = expert_w + (((size_t)e * CCH + o) * CCH + i) * 9;
#pragma unroll
            for (int r = 0; r < 9; ++r) acc[r] += cw * p[r];
        }
    }
#pragma unroll
    for (int r = 0; r < 9; ++r)
        wc[(((size_t)b * CCH + o) * 9 + r) * CCH + islot] = (__bf16)acc[r];
}

// ---------------------------------------------------------------------------
// Kernel 4: implicit-GEMM conv on v_wmma_f32_16x16x32_bf16.
// Wave tile: 32(M = one image row) x 32(N = 2 out-channel tiles).
// 4 accumulators; per K-step 4 b128-pair loads (2 A sharing one pointer via
// +16*192 imm offset, 2 B) feed 4 WMMAs -> 16 FLOP per L1 byte.
// Depth-2 software pipeline (3 slots) enforced with sched_barrier; the
// 54-step K-loop is force-fully-unrolled so all slot/array indices and load
// offsets are compile-time (no movrel, no runtime /6).
// Block = 8 waves sharing (b, npair) -> identical B lines (WGP$ reuse).
// ---------------------------------------------------------------------------
__global__ __launch_bounds__(256) void moe_gemm(
    const __bf16* __restrict__ xp, const __bf16* __restrict__ wc,
    const float* __restrict__ combine, const float* __restrict__ expert_b,
    float* __restrict__ out)
{
    const int wave = threadIdx.x >> 5;
    const int lane = threadIdx.x & 31;

    // block -> (b, npair, mgrp); wave -> image row (M-block of 32 positions)
    const int mgrp  = blockIdx.x % 4;
    const int npair = (blockIdx.x / 4) % 6;
    const int b     = blockIdx.x / 24;
    const int row   = mgrp * 8 + wave;       // 0..31

    const int l15 = lane & 15;
    const int hl  = lane >> 4;

    // A-fragment 0: positions (row, l15); fragment 1: (row, l15+16)
    unsigned aoff[9];
#pragma unroll
    for (int r = 0; r < 9; ++r) {
        const int ph = row + r / 3;          // padded row
        const int pw = l15 + r % 3;          // padded col (frag0)
        aoff[r] = (unsigned)((b * (PAD * PAD) + ph * PAD + pw) * CCH + hl * 16);
    }
    unsigned boff[2];
#pragma unroll
    for (int nn = 0; nn < 2; ++nn) {
        const int n = (npair * 2 + nn) * 16 + l15;
        boff[nn] = (unsigned)((b * CCH + n) * KTOT + hl * 16);
    }

    v8f acc[4];                              // [m*2 + nn]
#pragma unroll
    for (int q = 0; q < 4; ++q)
#pragma unroll
        for (int v = 0; v < 8; ++v) acc[q][v] = 0.f;

    // ---- 3-slot fragment buffers; preload steps 0 and 1 ----
    v16bf a0S[3], a1S[3], b0S[3], b1S[3];
#pragma unroll
    for (int p = 0; p < 2; ++p) {
        const int is = p * 32;               // r == 0 for steps 0,1
        a0S[p] = *reinterpret_cast<const v16bf*>(xp + aoff[0] + is);
        a1S[p] = *reinterpret_cast<const v16bf*>(xp + aoff[0] + is + 16 * CCH);
        b0S[p] = *reinterpret_cast<const v16bf*>(wc + boff[0] + p * 32);
        b1S[p] = *reinterpret_cast<const v16bf*>(wc + boff[1] + p * 32);
    }
    __builtin_amdgcn_sched_barrier(0);

#pragma clang loop unroll(full)
    for (int s = 0; s < NSTEP; ++s) {
        const int cur = s % 3;
        if (s + 2 < NSTEP) {                 // issue loads for step s+2
            const int nxt = (s + 2) % 3;
            const int r   = (s + 2) / 6;     // compile-time after full unroll
            const int is  = ((s + 2) % 6) * 32;
            a0S[nxt] = *reinterpret_cast<const v16bf*>(xp + aoff[r] + is);
            a1S[nxt] = *reinterpret_cast<const v16bf*>(xp + aoff[r] + is + 16 * CCH);
            b0S[nxt] = *reinterpret_cast<const v16bf*>(wc + boff[0] + (s + 2) * 32);
            b1S[nxt] = *reinterpret_cast<const v16bf*>(wc + boff[1] + (s + 2) * 32);
        }
        __builtin_amdgcn_sched_barrier(0x7); // only ALU may cross
        acc[0] = __builtin_amdgcn_wmma_f32_16x16x32_bf16(
            false, a0S[cur], false, b0S[cur], (short)0, acc[0], false, false);
        acc[1] = __builtin_amdgcn_wmma_f32_16x16x32_bf16(
            false, a0S[cur], false, b1S[cur], (short)0, acc[1], false, false);
        acc[2] = __builtin_amdgcn_wmma_f32_16x16x32_bf16(
            false, a1S[cur], false, b0S[cur], (short)0, acc[2], false, false);
        acc[3] = __builtin_amdgcn_wmma_f32_16x16x32_bf16(
            false, a1S[cur], false, b1S[cur], (short)0, acc[3], false, false);
        __builtin_amdgcn_sched_barrier(0x7);
    }

    // Epilogue: combined bias + vectorized stores. C/D layout: lane (n=l15,
    // M-half=hl), VGPR v -> M = hl*8 + v. Frag m covers cols m*16..m*16+15.
#pragma unroll
    for (int nn = 0; nn < 2; ++nn) {
        const int n = (npair * 2 + nn) * 16 + l15;
        float bias = 0.f;
#pragma unroll
        for (int e = 0; e < NEXP; ++e)
            bias += combine[b * NEXP + e] * expert_b[e * CCH + n];
#pragma unroll
        for (int m = 0; m < 2; ++m) {
            const v8f& a = acc[m * 2 + nn];
            float* op = out + (size_t)(b * CCH + n) * HWSZ
                            + row * 32 + m * 16 + hl * 8;
            float4 lo, hi;
            lo.x = a[0] + bias; lo.y = a[1] + bias;
            lo.z = a[2] + bias; lo.w = a[3] + bias;
            hi.x = a[4] + bias; hi.y = a[5] + bias;
            hi.z = a[6] + bias; hi.w = a[7] + bias;
            *reinterpret_cast<float4*>(op)     = lo;
            *reinterpret_cast<float4*>(op + 4) = hi;
        }
    }
}

// ---------------------------------------------------------------------------
extern "C" void kernel_launch(void* const* d_in, const int* in_sizes, int n_in,
                              void* d_out, int out_size, void* d_ws, size_t ws_size,
                              hipStream_t stream) {
    const float* x        = (const float*)d_in[0];  // [16,192,32,32]
    const float* gate_w   = (const float*)d_in[1];  // [192,8]
    const float* gate_b   = (const float*)d_in[2];  // [8]
    const float* expert_w = (const float*)d_in[3];  // [8,192,192,3,3]
    const float* expert_b = (const float*)d_in[4];  // [8,192]
    float* out = (float*)d_out;                     // [16,192,32,32]

    char* ws = (char*)d_ws;
    float*  combine = (float*)ws;                                        // 512 B
    __bf16* xpad    = (__bf16*)(ws + 512);                               // 7,102,464 B
    __bf16* wc      = (__bf16*)(ws + 512 + (size_t)BATCH * PAD * PAD * CCH * 2);

    moe_router<<<BATCH, 192, 0, stream>>>(x, gate_w, gate_b, combine);
    {
        const int total = BATCH * PAD * PAD * CCH;
        moe_xpad<<<(total + 255) / 256, 256, 0, stream>>>(x, xpad);
    }
    {
        const int total = BATCH * CCH * CCH;
        moe_wcomb<<<(total + 255) / 256, 256, 0, stream>>>(expert_w, combine, wc);
    }
    {
        // blocks: (b=16) x (npair=6) x (mgrp=4) = 384; 8 waves/block share B
        moe_gemm<<<BATCH * 6 * 4, 256, 0, stream>>>(xpad, wc, combine, expert_b, out);
    }
}
